// ProbSparseSelfAttention_20186346291965
// MI455X (gfx1250) — matrix operand order repaired and checked by reading, not compiled, not run on hardware
//
#include <hip/hip_runtime.h>
#include <hip/hip_bf16.h>

// ---------------------------------------------------------------------------
// ProbSparse self-attention for MI455X (gfx1250, wave32, WMMA).
// B=2, N=2048, C=1024, H=16, dk=64.  bf16 WMMA (16x16x32) with f32 accum.
// GEMMs use double-buffered async global->LDS copies (ASYNCcnt path).
// ---------------------------------------------------------------------------

typedef __attribute__((ext_vector_type(16))) __bf16 v16bf;
typedef __attribute__((ext_vector_type(8)))  float  v8f;
typedef __attribute__((ext_vector_type(4)))  float  f4raw;

union FragBF { f4raw f4[2]; v16bf v; };

__device__ __forceinline__ v8f wmma_bf16(v16bf a, v16bf b, v8f c) {
  // D = A(16x32) * B(32x16) + C, f32 accumulate
  return __builtin_amdgcn_wmma_f32_16x16x32_bf16(false, a, false, b,
                                                 (short)0, c, false, false);
}

__device__ __forceinline__ v8f v8f_zero() {
  v8f z = {0.f, 0.f, 0.f, 0.f, 0.f, 0.f, 0.f, 0.f};
  return z;
}

// LDS byte offset of a __shared__ object (generic -> AS(3) -> int)
__device__ __forceinline__ unsigned lds_off(const void* p) {
  return (unsigned)(unsigned long long)(__attribute__((address_space(3))) const void*)p;
}

// Async global->LDS 16-byte copy per lane (tracked by ASYNCcnt).
__device__ __forceinline__ void async_ld_b128(unsigned ldsOff, const void* g) {
  asm volatile("global_load_async_to_lds_b128 %0, %1, off"
               :: "v"(ldsOff), "v"(g) : "memory");
}
__device__ __forceinline__ void wait_async0() {
  asm volatile("s_wait_asynccnt 0x0" ::: "memory");
}
__device__ __forceinline__ void wait_xcnt0() {
  asm volatile("s_wait_xcnt 0x0" ::: "memory");
}

#define BB 2
#define NN 2048
#define CC 1024
#define HH 16
#define DK 64
#define BH (BB * HH)      // 32
#define SCALE 0.125f      // 1/sqrt(64)
#define NTOP 512          // N * 0.25

// ---------------------------------------------------------------------------
// K0: f32 -> bf16 conversion
// ---------------------------------------------------------------------------
__global__ __launch_bounds__(256) void cvt_bf16_kernel(const float* __restrict__ in,
                                                       __bf16* __restrict__ out,
                                                       long n) {
  long i = (long)blockIdx.x * 256 + threadIdx.x;
  long stride = (long)gridDim.x * 256;
  for (; i < n; i += stride) out[i] = (__bf16)in[i];
}

// ---------------------------------------------------------------------------
// K1/K6: 128x128-tile GEMM, C[m,n] = sum_k A[m,k]*Bw[n,k] + bias[n]
// Double-buffered async global->LDS tile staging overlapped with WMMA.
// MODE 0: scatter into Q [bh][n][dk], K [bh][n][dk], Vt [bh][dk][n] (bf16)
// MODE 1: f32 store to Yout[m*CC + n]
// 256 threads = 8 waves (4 M x 2 N), each wave: 32(M) x 64(N) = 2x4 frags.
// ---------------------------------------------------------------------------
template <int MODE>
__global__ __launch_bounds__(256) void gemm128_bf16_kernel(
    const __bf16* __restrict__ A, const __bf16* __restrict__ Bw,
    const float* __restrict__ bias,
    __bf16* __restrict__ Qb, __bf16* __restrict__ Kb, __bf16* __restrict__ Vtb,
    float* __restrict__ Yout) {
  constexpr int KD = 1024;
  constexpr int LS = 40;  // padded LDS row stride (bf16 elems)
  __shared__ __align__(16) __bf16 Asl[2][128 * LS];
  __shared__ __align__(16) __bf16 Bsl[2][128 * LS];

  const int t = threadIdx.x;
  const int lane = t & 31;
  const int w = t >> 5;
  const int wm = w >> 1;
  const int wn = w & 1;
  const int l16 = lane & 15;
  const int g = lane >> 4;
  const long mbase = (long)blockIdx.x * 128;
  const long nbase = (long)blockIdx.y * 128;

  // Each thread copies 2 x 16B for A and B per tile.
  const int row0 = t >> 2, seg0 = t & 3;
  const int row1 = (t + 256) >> 2, seg1 = (t + 256) & 3;

  auto issue_tile = [&](int buf, int kb) {
    async_ld_b128(lds_off(&Asl[buf][row0 * LS + seg0 * 8]),
                  &A[(mbase + row0) * KD + kb + seg0 * 8]);
    async_ld_b128(lds_off(&Bsl[buf][row0 * LS + seg0 * 8]),
                  &Bw[(nbase + row0) * KD + kb + seg0 * 8]);
    async_ld_b128(lds_off(&Asl[buf][row1 * LS + seg1 * 8]),
                  &A[(mbase + row1) * KD + kb + seg1 * 8]);
    async_ld_b128(lds_off(&Bsl[buf][row1 * LS + seg1 * 8]),
                  &Bw[(nbase + row1) * KD + kb + seg1 * 8]);
    wait_xcnt0();  // translations done -> safe to recycle address VGPRs
  };

  v8f acc[2][4];
#pragma unroll
  for (int i = 0; i < 2; ++i)
#pragma unroll
    for (int j = 0; j < 4; ++j) acc[i][j] = v8f_zero();

  issue_tile(0, 0);
  wait_async0();
  __syncthreads();

#pragma unroll 1
  for (int kb = 0; kb < KD; kb += 32) {
    const int cur = (kb >> 5) & 1;
    if (kb + 32 < KD) issue_tile(cur ^ 1, kb + 32);  // prefetch next tile

    FragBF af[2], bfr[4];
#pragma unroll
    for (int mi = 0; mi < 2; ++mi) {
      int r = wm * 32 + mi * 16 + l16;
      af[mi].f4[0] = *(const f4raw*)&Asl[cur][r * LS + g * 8];       // K=g*8..
      af[mi].f4[1] = *(const f4raw*)&Asl[cur][r * LS + 16 + g * 8];  // K=16+g*8..
    }
#pragma unroll
    for (int nj = 0; nj < 4; ++nj) {
      int r = wn * 64 + nj * 16 + l16;
      bfr[nj].f4[0] = *(const f4raw*)&Bsl[cur][r * LS + g * 8];       // K=g*8..
      bfr[nj].f4[1] = *(const f4raw*)&Bsl[cur][r * LS + 16 + g * 8];  // K=16+g*8..
    }
#pragma unroll
    for (int mi = 0; mi < 2; ++mi)
#pragma unroll
      for (int nj = 0; nj < 4; ++nj)
        acc[mi][nj] = wmma_bf16(af[mi].v, bfr[nj].v, acc[mi][nj]);

    wait_async0();     // next tile landed (per-wave counter)
    __syncthreads();   // visible to all waves
  }

#pragma unroll
  for (int mi = 0; mi < 2; ++mi)
#pragma unroll
    for (int nj = 0; nj < 4; ++nj)
#pragma unroll
      for (int r = 0; r < 8; ++r) {
        long m = mbase + wm * 32 + mi * 16 + r + 8 * g;   // D layout: row
        long n = nbase + wn * 64 + nj * 16 + l16;         // D layout: col
        float val = acc[mi][nj][r] + bias[n];
        if (MODE == 0) {
          int b = (int)(m >> 11), q = (int)(m & 2047);
          int s = (int)(n >> 10);
          int rem = (int)(n & 1023);
          int h = rem >> 6;
          int d = rem & 63;
          long bh = (long)b * HH + h;
          __bf16 bv = (__bf16)val;
          if (s == 0)      Qb[(bh * NN + q) * DK + d] = bv;
          else if (s == 1) Kb[(bh * NN + q) * DK + d] = bv;
          else             Vtb[(bh * DK + d) * NN + q] = bv;  // V transposed
        } else {
          Yout[m * CC + n] = val;
        }
      }
}

// ---------------------------------------------------------------------------
// K2: per-row stats of S = Q K^T (unmaterialized).
// rowmax[bh][q] = SCALE*max_k s, Mval = SCALE*(max - mean).
// One wave per 16 query rows; K-frags streamed from global (L2 resident).
// ---------------------------------------------------------------------------
__global__ __launch_bounds__(256) void score_stats_kernel(
    const __bf16* __restrict__ Qb, const __bf16* __restrict__ Kb,
    float* __restrict__ rowmax, float* __restrict__ Mval) {
  const int blk = blockIdx.x;
  const int bh = blk >> 4;
  const int qt = blk & 15;
  const int t = threadIdx.x;
  const int lane = t & 31;
  const int w = t >> 5;
  const int l16 = lane & 15;
  const int g = lane >> 4;
  const int q0 = qt * 128 + w * 16;

  const __bf16* Qp = Qb + (long)bh * NN * DK;
  const __bf16* Kp = Kb + (long)bh * NN * DK;

  FragBF aq[2];
  {
    long qr = (long)(q0 + l16) * DK;
    aq[0].f4[0] = *(const f4raw*)&Qp[qr + g * 8];
    aq[0].f4[1] = *(const f4raw*)&Qp[qr + 16 + g * 8];
    aq[1].f4[0] = *(const f4raw*)&Qp[qr + 32 + g * 8];
    aq[1].f4[1] = *(const f4raw*)&Qp[qr + 48 + g * 8];
  }

  float mx[8], sm[8];
#pragma unroll
  for (int r = 0; r < 8; ++r) { mx[r] = -3.4e38f; sm[r] = 0.f; }

#pragma unroll 1
  for (int kc = 0; kc < NN; kc += 16) {
    if (kc + 128 < NN)  // warm near cache a few tiles ahead (WGP scope)
      __builtin_prefetch(&Kp[(long)(kc + 128 + l16) * DK], 0, 3);
    long kr = (long)(kc + l16) * DK;
    FragBF b0, b1;
    b0.f4[0] = *(const f4raw*)&Kp[kr + g * 8];
    b0.f4[1] = *(const f4raw*)&Kp[kr + 16 + g * 8];
    b1.f4[0] = *(const f4raw*)&Kp[kr + 32 + g * 8];
    b1.f4[1] = *(const f4raw*)&Kp[kr + 48 + g * 8];
    v8f c = v8f_zero();
    c = wmma_bf16(aq[0].v, b0.v, c);   // d = 0..31
    c = wmma_bf16(aq[1].v, b1.v, c);   // d = 32..63
#pragma unroll
    for (int r = 0; r < 8; ++r) {
      mx[r] = fmaxf(mx[r], c[r]);
      sm[r] += c[r];
    }
  }
  // reduce across the 16 columns held by each half-wave
#pragma unroll
  for (int msk = 1; msk < 16; msk <<= 1)
#pragma unroll
    for (int r = 0; r < 8; ++r) {
      mx[r] = fmaxf(mx[r], __shfl_xor(mx[r], msk, 32));
      sm[r] += __shfl_xor(sm[r], msk, 32);
    }
  if (l16 == 0) {
#pragma unroll
    for (int r = 0; r < 8; ++r) {
      int q = q0 + r + 8 * g;
      rowmax[(long)bh * NN + q] = SCALE * mx[r];
      Mval[(long)bh * NN + q] = SCALE * (mx[r] - sm[r] * (1.0f / (float)NN));
    }
  }
}

// ---------------------------------------------------------------------------
// K3: per-(b,h) bitonic sort of M -> threshold = 512th largest value.
// ---------------------------------------------------------------------------
__global__ __launch_bounds__(256) void topk_thresh_kernel(
    const float* __restrict__ Mval, float* __restrict__ thresh) {
  __shared__ float s[NN];
  const int bh = blockIdx.x;
  const int t = threadIdx.x;
  for (int i = t; i < NN; i += 256) s[i] = Mval[(long)bh * NN + i];
  __syncthreads();
  for (int ksz = 2; ksz <= NN; ksz <<= 1) {
    for (int j = ksz >> 1; j > 0; j >>= 1) {
      for (int idx = t; idx < NN; idx += 256) {
        int ixj = idx ^ j;
        if (ixj > idx) {
          bool up = ((idx & ksz) == 0);
          float a = s[idx], b = s[ixj];
          if ((a > b) == up) { s[idx] = b; s[ixj] = a; }
        }
      }
      __syncthreads();
    }
  }
  if (t == 0) thresh[bh] = s[NN - NTOP];  // ascending sort
}

// ---------------------------------------------------------------------------
// K4: column means of V (output for fully-masked rows: softmax(const)=1/N).
// ---------------------------------------------------------------------------
__global__ __launch_bounds__(64) void vmean_kernel(const __bf16* __restrict__ Vtb,
                                                   float* __restrict__ vmean) {
  const int bh = blockIdx.x;
  const int d = threadIdx.x;  // 64 threads
  const __bf16* p = Vtb + ((long)bh * DK + d) * NN;
  float s = 0.f;
  for (int n = 0; n < NN; ++n) s += (float)p[n];
  vmean[bh * DK + d] = s * (1.0f / (float)NN);
}

// ---------------------------------------------------------------------------
// K5: attention output. Flash-style second pass with precomputed rowmax.
// Per wave: 16 q rows x full dk=64; loop over key tiles of 32.
// P goes D-layout -> LDS -> A-layout for the P*V WMMA.
// ---------------------------------------------------------------------------
__global__ __launch_bounds__(256) void attn_out_kernel(
    const __bf16* __restrict__ Qb, const __bf16* __restrict__ Kb,
    const __bf16* __restrict__ Vtb, const float* __restrict__ rowmax,
    const float* __restrict__ Mval, const float* __restrict__ thresh,
    const float* __restrict__ vmean, __bf16* __restrict__ OutP) {
  constexpr int PS = 40;  // padded stride for the 16x32 P tile
  __shared__ __align__(16) __bf16 Pl[8 * 16 * PS];

  const int blk = blockIdx.x;
  const int bh = blk >> 4;
  const int qt = blk & 15;
  const int b = bh >> 4;
  const int h = bh & 15;
  const int t = threadIdx.x;
  const int lane = t & 31;
  const int w = t >> 5;
  const int l16 = lane & 15;
  const int g = lane >> 4;
  const int q0 = qt * 128 + w * 16;

  const __bf16* Qp = Qb + (long)bh * NN * DK;
  const __bf16* Kp = Kb + (long)bh * NN * DK;
  const __bf16* Vp = Vtb + (long)bh * DK * NN;
  __bf16* Pw = &Pl[w * 16 * PS];

  FragBF aq[2];
  {
    long qr = (long)(q0 + l16) * DK;
    aq[0].f4[0] = *(const f4raw*)&Qp[qr + g * 8];
    aq[0].f4[1] = *(const f4raw*)&Qp[qr + 16 + g * 8];
    aq[1].f4[0] = *(const f4raw*)&Qp[qr + 32 + g * 8];
    aq[1].f4[1] = *(const f4raw*)&Qp[qr + 48 + g * 8];
  }
  float rm[8], ps[8];
#pragma unroll
  for (int r = 0; r < 8; ++r) {
    rm[r] = rowmax[(long)bh * NN + q0 + r + 8 * g];
    ps[r] = 0.f;
  }
  v8f o[4];
#pragma unroll
  for (int j = 0; j < 4; ++j) o[j] = v8f_zero();

#pragma unroll 1
  for (int kc = 0; kc < NN; kc += 32) {
    if (kc + 128 < NN)
      __builtin_prefetch(&Kp[(long)(kc + 128 + l16) * DK], 0, 3);
    // S tile (16q x 32kc) as two 16x16 frags; p = exp(scale*s - rowmax)
#pragma unroll
    for (int f = 0; f < 2; ++f) {
      long kr = (long)(kc + f * 16 + l16) * DK;
      FragBF b0, b1;
      b0.f4[0] = *(const f4raw*)&Kp[kr + g * 8];
      b0.f4[1] = *(const f4raw*)&Kp[kr + 16 + g * 8];
      b1.f4[0] = *(const f4raw*)&Kp[kr + 32 + g * 8];
      b1.f4[1] = *(const f4raw*)&Kp[kr + 48 + g * 8];
      v8f c = v8f_zero();
      c = wmma_bf16(aq[0].v, b0.v, c);
      c = wmma_bf16(aq[1].v, b1.v, c);
#pragma unroll
      for (int r = 0; r < 8; ++r) {
        float p = __expf(SCALE * c[r] - rm[r]);
        ps[r] += p;
        Pw[(r + 8 * g) * PS + f * 16 + l16] = (__bf16)p;  // D -> LDS
      }
    }
    // P as A-frag (16x32), V columns as B-frags (32 x 16d each)
    FragBF ap;
    ap.f4[0] = *(const f4raw*)&Pw[l16 * PS + g * 8];
    ap.f4[1] = *(const f4raw*)&Pw[l16 * PS + 16 + g * 8];
#pragma unroll
    for (int j = 0; j < 4; ++j) {
      long vr = (long)(j * 16 + l16) * NN + kc;
      FragBF bv;
      bv.f4[0] = *(const f4raw*)&Vp[vr + g * 8];
      bv.f4[1] = *(const f4raw*)&Vp[vr + 16 + g * 8];
      o[j] = wmma_bf16(ap.v, bv.v, o[j]);
    }
  }

  // row sums of p across the 16 columns per half-wave (all lanes get result)
#pragma unroll
  for (int msk = 1; msk < 16; msk <<= 1)
#pragma unroll
    for (int r = 0; r < 8; ++r) ps[r] += __shfl_xor(ps[r], msk, 32);

#pragma unroll
  for (int r = 0; r < 8; ++r) {
    int q = q0 + r + 8 * g;
    bool kept = Mval[(long)bh * NN + q] >= thresh[bh];
    float inv = 1.0f / ps[r];
#pragma unroll
    for (int j = 0; j < 4; ++j) {
      int d = j * 16 + l16;
      float val = kept ? o[j][r] * inv : vmean[bh * DK + d];
      OutP[((long)(b * NN + q)) * CC + h * DK + d] = (__bf16)val;
    }
  }
}

// ---------------------------------------------------------------------------
// Launch
// ---------------------------------------------------------------------------
extern "C" void kernel_launch(void* const* d_in, const int* in_sizes, int n_in,
                              void* d_out, int out_size, void* d_ws, size_t ws_size,
                              hipStream_t stream) {
  const float* x    = (const float*)d_in[0];
  const float* Wqkv = (const float*)d_in[1];
  const float* bqkv = (const float*)d_in[2];
  const float* Wout = (const float*)d_in[3];
  const float* bout = (const float*)d_in[4];
  (void)in_sizes; (void)n_in; (void)out_size; (void)ws_size;

  char* ws = (char*)d_ws;
  size_t cur = 0;
  auto take = [&](size_t bytes) -> char* {
    char* p = ws + cur;
    cur += (bytes + 255) & ~(size_t)255;
    return p;
  };
  __bf16* Xbf  = (__bf16*)take((size_t)BB * NN * CC * 2);
  __bf16* Wqb  = (__bf16*)take((size_t)3 * CC * CC * 2);
  __bf16* Wob  = (__bf16*)take((size_t)CC * CC * 2);
  __bf16* Qb   = (__bf16*)take((size_t)BH * NN * DK * 2);
  __bf16* Kb   = (__bf16*)take((size_t)BH * NN * DK * 2);
  __bf16* Vtb  = (__bf16*)take((size_t)BH * NN * DK * 2);
  float*  rmax = (float*)take((size_t)BH * NN * 4);
  float*  Mv   = (float*)take((size_t)BH * NN * 4);
  float*  thr  = (float*)take((size_t)BH * 4);
  float*  vmn  = (float*)take((size_t)BH * DK * 4);
  __bf16* OutP = (__bf16*)take((size_t)BB * NN * CC * 2);

  cvt_bf16_kernel<<<2048, 256, 0, stream>>>(x, Xbf, (long)BB * NN * CC);
  cvt_bf16_kernel<<<2048, 256, 0, stream>>>(Wqkv, Wqb, (long)3 * CC * CC);
  cvt_bf16_kernel<<<1024, 256, 0, stream>>>(Wout, Wob, (long)CC * CC);

  // QKV projection: (4096 x 1024) @ (3072 x 1024)^T
  gemm128_bf16_kernel<0><<<dim3(32, 24), 256, 0, stream>>>(
      Xbf, Wqb, bqkv, Qb, Kb, Vtb, nullptr);

  // per-row score stats (streams S without materializing it)
  score_stats_kernel<<<BH * 16, 256, 0, stream>>>(Qb, Kb, rmax, Mv);

  // top-25% threshold per (b,h)
  topk_thresh_kernel<<<BH, 256, 0, stream>>>(Mv, thr);

  // V column means for masked rows
  vmean_kernel<<<BH, 64, 0, stream>>>(Vtb, vmn);

  // attention output (flash-style second pass)
  attn_out_kernel<<<BH * 16, 256, 0, stream>>>(Qb, Kb, Vtb, rmax, Mv, thr, vmn, OutP);

  // output projection: (4096 x 1024) @ (1024 x 1024)^T -> f32 d_out
  gemm128_bf16_kernel<1><<<dim3(32, 8), 256, 0, stream>>>(
      OutP, Wob, bout, nullptr, nullptr, nullptr, (float*)d_out);
}
